// ViSiLHead_41162966565490
// MI455X (gfx1250) — compile-verified, matrix operand order checked by reading, not run on hardware
//
#include <hip/hip_runtime.h>
#include <hip/hip_bf16.h>

// ---------------- constants ----------------
constexpr int Bn  = 4;      // batch
constexpr int TQ  = 256;    // query frames
constexpr int TT  = 256;    // target frames
constexpr int R   = 9;      // regions
constexpr int D   = 3840;   // feature dim
constexpr int MROWS = TQ * R;          // 2304 (== TT*R)
constexpr int KDIM  = D;               // 3840
constexpr int KTILE = 64;              // K-tile per LDS buffer
constexpr int KTILES = KDIM / KTILE;   // 60
constexpr int LDS_STRIDE = KTILE + 8;  // 72 halfs = 144B (16B multiple)

typedef __attribute__((ext_vector_type(16))) __bf16 v16bf;
typedef __attribute__((ext_vector_type(8)))  float  v8f;

// ---------------- 1) L2 normalize + cast to bf16 ----------------
__global__ __launch_bounds__(256)
void l2norm_bf16_kernel(const float* __restrict__ in, __bf16* __restrict__ out) {
    const int row = blockIdx.x;
    const int tid = threadIdx.x;
    const float* rp = in + (size_t)row * D;

    float ss = 0.0f;
    for (int c = tid; c < D; c += 256) {
        float v = rp[c];
        ss += v * v;
    }
    __shared__ float red[256];
    red[tid] = ss;
    __syncthreads();
    #pragma unroll
    for (int off = 128; off > 0; off >>= 1) {
        if (tid < off) red[tid] += red[tid + off];
        __syncthreads();
    }
    __shared__ float inv_s;
    if (tid == 0) {
        float nrm = sqrtf(red[0]);
        inv_s = 1.0f / fmaxf(nrm, 1e-12f);
    }
    __syncthreads();
    const float inv = inv_s;
    __bf16* op = out + (size_t)row * D;
    for (int c = tid; c < D; c += 256) {
        op[c] = (__bf16)(rp[c] * inv);
    }
}

// ---------------- 2) WMMA GEMM: S[b] = Qn[b] (2304xK) * Tn[b]^T ----------------
// Block tile 128x128, 8 waves laid out 4(M) x 2(N); wave tile 32x64 = 2x4 WMMA tiles.
// Operands staged via CDNA5 async global->LDS copies (ASYNCcnt), double-buffered,
// K-tile 64 split into two 16x16x32 bf16 WMMA k-steps.
union FragBF { v16bf v; uint4 q[2]; };

__global__ __launch_bounds__(256)
void gemm_qt_kernel(const __bf16* __restrict__ Q, const __bf16* __restrict__ T,
                    float* __restrict__ S) {
    const int b     = blockIdx.z;
    const int mTile = blockIdx.y;   // 0..17
    const int nTile = blockIdx.x;   // 0..17

    const int tid  = threadIdx.x;
    const int wave = tid >> 5;
    const int lane = tid & 31;
    const int lr   = lane & 15;     // M (A) / N (B) index within tile
    const int hi   = lane >> 4;     // K-half select

    const int wm = wave & 3;        // 0..3 -> 32-row strip
    const int wn = wave >> 2;       // 0..1 -> 64-col strip

    // ping-pong LDS tiles: stride 72 halfs = 144B (multiple of 16B)
    __shared__ alignas(16) __bf16 As[2][128][LDS_STRIDE];
    __shared__ alignas(16) __bf16 Bs[2][128][LDS_STRIDE];

    const __bf16* Qb = Q + (size_t)b * MROWS * KDIM + (size_t)(mTile * 128) * KDIM;
    const __bf16* Tb = T + (size_t)b * MROWS * KDIM + (size_t)(nTile * 128) * KDIM;

    // Issue one tile-fill: 128 rows x 64 halfs per operand = 1024 x 16B chunks,
    // 256 threads x 4 chunks each, via async global->LDS b128 copies (GVS mode).
    auto fill = [&](int buf, int k0) {
        #pragma unroll
        for (int l = 0; l < 4; ++l) {
            const int lin = tid * 4 + l;          // 0..1023
            const int row = lin >> 3;             // 0..127
            const int kc  = lin & 7;              // 0..7 -> 8-half chunk
            const unsigned goff = (unsigned)(row * (KDIM * 2) + (k0 + kc * 8) * 2);
            const unsigned aoff = (unsigned)(uintptr_t)&As[buf][row][kc * 8];
            const unsigned boff = (unsigned)(uintptr_t)&Bs[buf][row][kc * 8];
            asm volatile("global_load_async_to_lds_b128 %0, %1, %2"
                         :: "v"(aoff), "v"(goff), "s"(Qb) : "memory");
            asm volatile("global_load_async_to_lds_b128 %0, %1, %2"
                         :: "v"(boff), "v"(goff), "s"(Tb) : "memory");
        }
    };

    v8f acc[2][4];
    const v8f vzero = {0.f, 0.f, 0.f, 0.f, 0.f, 0.f, 0.f, 0.f};
    #pragma unroll
    for (int mi = 0; mi < 2; ++mi)
        #pragma unroll
        for (int ni = 0; ni < 4; ++ni)
            acc[mi][ni] = vzero;

    // preload both buffers
    fill(0, 0);
    fill(1, KTILE);

    for (int kt = 0; kt < KTILES; ++kt) {
        const int cur = kt & 1;
        // Async loads complete in issue order: <=8 outstanding means the
        // current buffer's 8 copies are done; last tile needs full drain.
        if (kt < KTILES - 1) {
            asm volatile("s_wait_asynccnt 0x8" ::: "memory");
        } else {
            asm volatile("s_wait_asynccnt 0x0" ::: "memory");
        }
        __syncthreads();   // current buffer visible to all waves

        #pragma unroll
        for (int s = 0; s < 2; ++s) {         // two 32-wide k-steps
            FragBF a[2], bfr[4];
            #pragma unroll
            for (int mi = 0; mi < 2; ++mi) {
                const int row = wm * 32 + mi * 16 + lr;
                a[mi].q[0] = *(const uint4*)&As[cur][row][s * 32 + hi * 8];
                a[mi].q[1] = *(const uint4*)&As[cur][row][s * 32 + 16 + hi * 8];
            }
            #pragma unroll
            for (int ni = 0; ni < 4; ++ni) {
                const int row = wn * 64 + ni * 16 + lr;
                bfr[ni].q[0] = *(const uint4*)&Bs[cur][row][s * 32 + hi * 8];
                bfr[ni].q[1] = *(const uint4*)&Bs[cur][row][s * 32 + 16 + hi * 8];
            }
            #pragma unroll
            for (int mi = 0; mi < 2; ++mi)
                #pragma unroll
                for (int ni = 0; ni < 4; ++ni)
                    acc[mi][ni] = __builtin_amdgcn_wmma_f32_16x16x32_bf16(
                        false, a[mi].v, false, bfr[ni].v, (short)0, acc[mi][ni], false, false);
        }

        __syncthreads();   // everyone done reading 'cur' before refilling it
        if (kt + 2 < KTILES) fill(cur, (kt + 2) * KTILE);
    }

    // store: C/D layout — VGPR v holds M = v + 8*hi, N = lr
    float* Sb = S + (size_t)b * MROWS * MROWS;
    const int base_m = mTile * 128 + wm * 32;
    const int base_n = nTile * 128 + wn * 64;
    #pragma unroll
    for (int mi = 0; mi < 2; ++mi) {
        #pragma unroll
        for (int ni = 0; ni < 4; ++ni) {
            #pragma unroll
            for (int v = 0; v < 8; ++v) {
                const int rrow = base_m + mi * 16 + v + 8 * hi;
                const int ccol = base_n + ni * 16 + lr;
                Sb[(size_t)rrow * MROWS + ccol] = acc[mi][ni][v];
            }
        }
    }
}

// ---------------- 3) f2f[b,i,j] = mean_o max_p S[b, i*9+o, j*9+p] ----------------
__global__ __launch_bounds__(256)
void f2f_reduce_kernel(const float* __restrict__ S, float* __restrict__ f2f) {
    const int idx = blockIdx.x * 256 + threadIdx.x;   // B*TQ*TT = 262144
    const int b = idx >> 16;
    const int i = (idx >> 8) & 255;
    const int j = idx & 255;
    const float* Sb = S + (size_t)b * MROWS * MROWS;
    float sum = 0.0f;
    #pragma unroll
    for (int o = 0; o < R; ++o) {
        const float* rp = Sb + (size_t)(i * R + o) * MROWS + j * R;
        float m = rp[0];
        #pragma unroll
        for (int p = 1; p < R; ++p) m = fmaxf(m, rp[p]);
        sum += m;
    }
    f2f[idx] = sum * (1.0f / R);
}

// ---------------- 4) generic 3x3 conv (pad=1) + relu ----------------
__global__ __launch_bounds__(256)
void conv3x3_relu_kernel(const float* __restrict__ in, const float* __restrict__ w,
                         const float* __restrict__ bias, float* __restrict__ out,
                         int Cin, int Cout, int H, int W, int total) {
    const int idx = blockIdx.x * 256 + threadIdx.x;
    if (idx >= total) return;
    const int x  = idx % W;
    const int y  = (idx / W) % H;
    const int co = (idx / (W * H)) % Cout;
    const int b  = idx / (W * H * Cout);

    float acc = bias[co];
    for (int ci = 0; ci < Cin; ++ci) {
        const float* ip = in + (((size_t)b * Cin + ci) * H) * W;
        const float* wp = w + ((size_t)co * Cin + ci) * 9;
        #pragma unroll
        for (int dy = 0; dy < 3; ++dy) {
            const int yy = y + dy - 1;
            if (yy < 0 || yy >= H) continue;
            #pragma unroll
            for (int dx = 0; dx < 3; ++dx) {
                const int xx = x + dx - 1;
                if (xx < 0 || xx >= W) continue;
                acc += ip[(size_t)yy * W + xx] * wp[dy * 3 + dx];
            }
        }
    }
    out[idx] = fmaxf(acc, 0.0f);
}

// ---------------- 5) 2x2 maxpool ----------------
__global__ __launch_bounds__(256)
void maxpool2_kernel(const float* __restrict__ in, float* __restrict__ out,
                     int C, int H, int W, int total) {
    const int idx = blockIdx.x * 256 + threadIdx.x;   // over B*C*(H/2)*(W/2)
    if (idx >= total) return;
    const int Ho = H >> 1, Wo = W >> 1;
    const int x  = idx % Wo;
    const int y  = (idx / Wo) % Ho;
    const int bc = idx / (Wo * Ho);
    const float* ip = in + ((size_t)bc * H + y * 2) * W + x * 2;
    float m = fmaxf(fmaxf(ip[0], ip[1]), fmaxf(ip[W], ip[W + 1]));
    out[idx] = m;
}

// ---------------- 6) final 1x1 conv (128->1) + clip ----------------
__global__ __launch_bounds__(256)
void conv1x1_clip_kernel(const float* __restrict__ in, const float* __restrict__ wf,
                         const float* __restrict__ bf, float* __restrict__ out,
                         int total) {
    const int idx = blockIdx.x * 256 + threadIdx.x;   // over B*64*64
    if (idx >= total) return;
    const int hw = idx & 4095;          // 64*64
    const int b  = idx >> 12;
    const float* ip = in + (size_t)b * 128 * 4096 + hw;
    float acc = bf[0];
    for (int ci = 0; ci < 128; ++ci)
        acc += ip[(size_t)ci * 4096] * wf[ci];
    out[idx] = fminf(fmaxf(acc, -1.0f), 1.0f);
}

// ---------------- 7) out[b] = mean_h max_w xf[b,h,w] ----------------
__global__ void final_reduce_kernel(const float* __restrict__ xf, float* __restrict__ out) {
    const int b = blockIdx.x;
    const int h = threadIdx.x;          // 64
    const float* rp = xf + ((size_t)b * 64 + h) * 64;
    float m = rp[0];
    #pragma unroll
    for (int w = 1; w < 64; ++w) m = fmaxf(m, rp[w]);
    __shared__ float s[64];
    s[h] = m;
    __syncthreads();
    #pragma unroll
    for (int off = 32; off > 0; off >>= 1) {
        if (h < off) s[h] += s[h + off];
        __syncthreads();
    }
    if (h == 0) out[b] = s[0] * (1.0f / 64.0f);
}

// ---------------- host launch ----------------
extern "C" void kernel_launch(void* const* d_in, const int* in_sizes, int n_in,
                              void* d_out, int out_size, void* d_ws, size_t ws_size,
                              hipStream_t stream) {
    const float* query  = (const float*)d_in[0];
    const float* target = (const float*)d_in[1];
    const float* w1 = (const float*)d_in[2];
    const float* b1 = (const float*)d_in[3];
    const float* w2 = (const float*)d_in[4];
    const float* b2 = (const float*)d_in[5];
    const float* w3 = (const float*)d_in[6];
    const float* b3 = (const float*)d_in[7];
    const float* wf = (const float*)d_in[8];
    const float* bf = (const float*)d_in[9];
    float* out = (float*)d_out;

    char* ws = (char*)d_ws;
    size_t off = 0;
    auto carve = [&](size_t bytes) -> char* {
        char* p = ws + off;
        off += (bytes + 255) & ~(size_t)255;
        return p;
    };

    const size_t opBytes = (size_t)Bn * MROWS * KDIM * sizeof(__bf16);   // 70.8 MB
    __bf16* qn  = (__bf16*)carve(opBytes);
    __bf16* tn  = (__bf16*)carve(opBytes);
    float*  S   = (float*)carve((size_t)Bn * MROWS * MROWS * sizeof(float)); // 84.9 MB
    float*  f2f = (float*)carve((size_t)Bn * TQ * TT * sizeof(float));
    float*  x1  = (float*)carve((size_t)Bn * 32 * 256 * 256 * sizeof(float));
    float*  p1  = (float*)carve((size_t)Bn * 32 * 128 * 128 * sizeof(float));
    float*  x2  = (float*)carve((size_t)Bn * 64 * 128 * 128 * sizeof(float));
    float*  p2  = (float*)carve((size_t)Bn * 64 * 64 * 64 * sizeof(float));
    float*  x3  = (float*)carve((size_t)Bn * 128 * 64 * 64 * sizeof(float));
    float*  xf  = (float*)carve((size_t)Bn * 64 * 64 * sizeof(float));

    // 1) normalize + cast
    l2norm_bf16_kernel<<<Bn * TQ * R, 256, 0, stream>>>(query, qn);
    l2norm_bf16_kernel<<<Bn * TT * R, 256, 0, stream>>>(target, tn);

    // 2) WMMA GEMM (async-copy double-buffered)
    gemm_qt_kernel<<<dim3(MROWS / 128, MROWS / 128, Bn), 256, 0, stream>>>(qn, tn, S);

    // 3) region max/mean reduce
    f2f_reduce_kernel<<<(Bn * TQ * TT) / 256, 256, 0, stream>>>(S, f2f);

    // 4) CNN chain
    {
        int total = Bn * 32 * 256 * 256;
        conv3x3_relu_kernel<<<(total + 255) / 256, 256, 0, stream>>>(f2f, w1, b1, x1, 1, 32, 256, 256, total);
    }
    {
        int total = Bn * 32 * 128 * 128;
        maxpool2_kernel<<<(total + 255) / 256, 256, 0, stream>>>(x1, p1, 32, 256, 256, total);
    }
    {
        int total = Bn * 64 * 128 * 128;
        conv3x3_relu_kernel<<<(total + 255) / 256, 256, 0, stream>>>(p1, w2, b2, x2, 32, 64, 128, 128, total);
    }
    {
        int total = Bn * 64 * 64 * 64;
        maxpool2_kernel<<<(total + 255) / 256, 256, 0, stream>>>(x2, p2, 64, 128, 128, total);
    }
    {
        int total = Bn * 128 * 64 * 64;
        conv3x3_relu_kernel<<<(total + 255) / 256, 256, 0, stream>>>(p2, w3, b3, x3, 64, 128, 64, 64, total);
    }
    {
        int total = Bn * 64 * 64;
        conv1x1_clip_kernel<<<(total + 255) / 256, 256, 0, stream>>>(x3, wf, bf, xf, total);
    }
    final_reduce_kernel<<<Bn, 64, 0, stream>>>(xf, out);
}
